// MultiHeadAttention_26843545600216
// MI455X (gfx1250) — matrix operand order repaired and checked by reading, not compiled, not run on hardware
//
#include <hip/hip_runtime.h>
#include <hip/hip_bf16.h>

// ---------------------------------------------------------------------------
// MI455X (gfx1250) causal MHA: bf16 WMMA + Tensor Data Mover double buffering.
// B=4, T=2048, C=1024, H=16, D=64.
// ---------------------------------------------------------------------------

#define B_ 4
#define T_ 2048
#define C_ 1024
#define H_ 16
#define D_ 64

typedef __attribute__((ext_vector_type(16))) __bf16 bf16x16;
typedef __attribute__((ext_vector_type(8)))  __bf16 bf16x8;
typedef __attribute__((ext_vector_type(8)))  float  f32x8;
typedef unsigned int u32x4 __attribute__((ext_vector_type(4)));
typedef int          i32x4 __attribute__((ext_vector_type(4)));
typedef int          i32x8 __attribute__((ext_vector_type(8)));

// ---- WMMA wrapper ---------------------------------------------------------
__device__ __forceinline__ f32x8 wmma_bf16(bf16x16 a, bf16x16 b, f32x8 c) {
  return __builtin_amdgcn_wmma_f32_16x16x32_bf16(
      false, a, false, b, (short)0, c, false, false);
}

// ---- A-operand fragment (16x32 MxK), rows K-contiguous --------------------
// lanes 0-15: M=lane, K [0..7],[16..23]; lanes 16-31: M=lane-16, K [8..15],[24..31]
__device__ __forceinline__ bf16x16 load_frag_a(const __bf16* p0, int rowStride,
                                               int m0, int k0, int lane) {
  const int row = m0 + (lane & 15);
  const int kb  = (lane < 16) ? 0 : 8;
  const __bf16* p = p0 + row * rowStride + k0 + kb;
  bf16x8 lo = *(const bf16x8*)(p);
  bf16x8 hi = *(const bf16x8*)(p + 16);
  bf16x16 r;
#pragma unroll
  for (int i = 0; i < 8; ++i) { r[i] = lo[i]; r[i + 8] = hi[i]; }
  return r;
}

// ---- B-operand fragment (32x16 KxN) from N-major K-contiguous tile --------
// lanes 0-15: N=lane, K [0..7],[16..23]; lanes 16-31: N=lane-16, K [8..15],[24..31]
__device__ __forceinline__ bf16x16 load_frag_b(const __bf16* pT, int rowStride,
                                               int n0, int k0, int lane) {
  const int col = n0 + (lane & 15);
  const int kb  = (lane < 16) ? 0 : 8;
  const __bf16* p = pT + col * rowStride + k0 + kb;
  bf16x8 lo = *(const bf16x8*)(p);
  bf16x8 hi = *(const bf16x8*)(p + 16);
  bf16x16 r;
#pragma unroll
  for (int i = 0; i < 8; ++i) { r[i] = lo[i]; r[i + 8] = hi[i]; }
  return r;
}

// ---- Tensor Data Mover: build D# and issue tensor_load_to_lds -------------
// data_size=2B; pad_enable with padI (interval code: DWORDs=2<<code) and
// padA (amount code: DWORDs=code+1).  2D when tile2==0, else 3D
// (z-stride = stride1, per D# addressing: z uses tensor_dim1_stride).
__device__ __forceinline__ void tdm_load(uint32_t lds_off, const void* gp,
                                         uint32_t tile0, uint32_t tile1,
                                         uint32_t tile2,
                                         uint64_t stride0, uint64_t stride1,
                                         uint32_t padI, uint32_t padA) {
  const uint64_t ga = (uint64_t)(uintptr_t)gp;
  u32x4 g0; i32x8 g1; i32x4 g2; i32x4 g3; i32x8 g4;
  g0[0] = 1u;                                   // count=1, load, user mode
  g0[1] = lds_off;                              // LDS byte address
  g0[2] = (uint32_t)ga;                         // global_addr[31:0]
  g0[3] = ((uint32_t)(ga >> 32) & 0x01FFFFFFu) | (2u << 30);  // [56:32] | type=2
  const uint32_t td0 = tile0, td1 = tile1;      // tensor dims = tile extents
  g1[0] = (int)((1u << 16) | (1u << 20) | (padI << 22) | (padA << 25));
  g1[1] = (int)((td0 & 0xFFFFu) << 16);
  g1[2] = (int)(((td0 >> 16) & 0xFFFFu) | ((td1 & 0xFFFFu) << 16));
  g1[3] = (int)(((td1 >> 16) & 0xFFFFu) | (tile0 << 16));
  g1[4] = (int)((tile1 & 0xFFFFu) | (tile2 << 16));
  g1[5] = (int)(uint32_t)stride0;
  g1[6] = (int)(((uint32_t)(stride0 >> 32) & 0xFFFFu) |
                ((uint32_t)(stride1 & 0xFFFFu) << 16));
  g1[7] = (int)(uint32_t)(stride1 >> 16);
  g2[0] = (int)tile2; g2[1] = 0; g2[2] = 0; g2[3] = 0;   // tensor_dim2
  g3[0] = 0; g3[1] = 0; g3[2] = 0; g3[3] = 0;
#pragma unroll
  for (int i = 0; i < 8; ++i) g4[i] = 0;
  __builtin_amdgcn_tensor_load_to_lds(g0, g1, g2, g3, g4, 0);
}

__device__ __forceinline__ uint32_t lds_off_of(const void* p) {
  return (uint32_t)(uintptr_t)p;   // generic LDS addr: offset in low 32 bits
}

// ===========================================================================
// Kernel 0a: x f32 -> bf16 (flat copy).
// ===========================================================================
__global__ __launch_bounds__(256)
void cvt_x_kernel(const float* __restrict__ x, __bf16* __restrict__ xb) {
  const size_t i = ((size_t)blockIdx.x * 256 + threadIdx.x) * 8;
  float4 a = *(const float4*)(x + i);
  float4 b = *(const float4*)(x + i + 4);
  bf16x8 o;
  o[0]=(__bf16)a.x; o[1]=(__bf16)a.y; o[2]=(__bf16)a.z; o[3]=(__bf16)a.w;
  o[4]=(__bf16)b.x; o[5]=(__bf16)b.y; o[6]=(__bf16)b.z; o[7]=(__bf16)b.w;
  *(bf16x8*)(xb + i) = o;
}

// ===========================================================================
// Kernel 0b: W[h][c][d] f32 -> wT[m][h][d][c] bf16 (transpose via LDS).
// Grid (C/64, H, 3), 256 threads.
// ===========================================================================
__global__ __launch_bounds__(256)
void cvt_w_kernel(const float* __restrict__ Wq, const float* __restrict__ Wk,
                  const float* __restrict__ Wv, __bf16* __restrict__ wT) {
  const int c0 = blockIdx.x * 64, h = blockIdx.y, m = blockIdx.z;
  const float* W = (m == 0) ? Wq : (m == 1) ? Wk : Wv;
  __shared__ __bf16 lds[64 * 72];
  for (int i = threadIdx.x; i < 64 * 64; i += 256) {
    const int c = i >> 6, d = i & 63;                       // coalesced over d
    lds[d * 72 + c] = (__bf16)W[((size_t)h * C_ + c0 + c) * D_ + d];
  }
  __syncthreads();
  for (int i = threadIdx.x; i < 512; i += 256) {
    const int d = i >> 3, c8 = (i & 7) * 8;
    *(bf16x8*)&wT[(((size_t)m * H_ + h) * D_ + d) * C_ + c0 + c8] =
        *(const bf16x8*)&lds[d * 72 + c8];
  }
}

// ===========================================================================
// Kernel 1: QKV projection with TDM double buffering.
// xb: [B*T, C] bf16 ; wT: [3,H,D,C] bf16.
// out q,k: [B,H,T,D] bf16 ; v: [B,H,D,T] bf16 (transposed).
// Grid (T/128, H, B), 256 threads (8 waves).
// ===========================================================================
#define XS 40   // 32 + 4-DWORD TDM pad
#define XTILE (128 * XS)          // elements per x buffer
#define WTILE (3 * 64 * XS)       // elements per W buffer (3 planes)
#define WPLANE (64 * XS)

__global__ __launch_bounds__(256)
void qkv_proj_kernel(const __bf16* __restrict__ xb, const __bf16* __restrict__ wT,
                     __bf16* __restrict__ qO, __bf16* __restrict__ kO,
                     __bf16* __restrict__ vTO) {
  const int tile = blockIdx.x, h = blockIdx.y, b = blockIdx.z;
  const int tid = threadIdx.x;
  const int wave = tid >> 5, lane = tid & 31;
  const int half = lane >> 4, nl = lane & 15;
  const int t0 = tile * 128;

  __shared__ __align__(16) char smem[(2 * XTILE + 2 * WTILE) * 2];
  __bf16* ldsX = (__bf16*)smem;                    // 2 x XTILE
  __bf16* ldsW = (__bf16*)smem + 2 * XTILE;        // 2 x WTILE
  __bf16* ldsVt = (__bf16*)smem;                   // overlay, used after loop

  const __bf16* xrow = xb + ((size_t)(b * T_ + t0)) * C_;
  const __bf16* wrow = wT + (size_t)h * D_ * C_;   // plane stride = H*D*C

  f32x8 accQ[4], accK[4], accV[4];
#pragma unroll
  for (int i = 0; i < 4; ++i) {
    accQ[i] = (f32x8)(0.0f); accK[i] = (f32x8)(0.0f); accV[i] = (f32x8)(0.0f);
  }

  // Prologue: DMA chunk 0 into buffer 0 (wave 0 only; EXEC ignored by TDM).
  if (wave == 0) {
    tdm_load(lds_off_of(ldsX), xrow, 32, 128, 0, C_, 0, /*padI=*/3, /*padA=*/3);
    tdm_load(lds_off_of(ldsW), wrow, 32, 64, 3, C_, (uint64_t)H_ * D_ * C_, 3, 3);
  }

  for (int ic = 0; ic < C_ / 32; ++ic) {
    if (wave == 0) {
      if (ic + 1 < C_ / 32) {
        const int c1 = (ic + 1) * 32;
        const int nb = (ic + 1) & 1;
        tdm_load(lds_off_of(ldsX + nb * XTILE), xrow + c1, 32, 128, 0, C_, 0, 3, 3);
        tdm_load(lds_off_of(ldsW + nb * WTILE), wrow + c1, 32, 64, 3, C_,
                 (uint64_t)H_ * D_ * C_, 3, 3);
        __builtin_amdgcn_s_wait_tensorcnt(2);   // current pair done
      } else {
        __builtin_amdgcn_s_wait_tensorcnt(0);
      }
    }
    __syncthreads();                            // buffer ic&1 ready for all

    const __bf16* Xb = ldsX + (ic & 1) * XTILE;
    const __bf16* Wb = ldsW + (ic & 1) * WTILE;
    const bf16x16 a = load_frag_a(Xb, XS, wave * 16, 0, lane);
#pragma unroll
    for (int nt = 0; nt < 4; ++nt) {
      accQ[nt] = wmma_bf16(a, load_frag_b(Wb,              XS, nt * 16, 0, lane), accQ[nt]);
      accK[nt] = wmma_bf16(a, load_frag_b(Wb + WPLANE,     XS, nt * 16, 0, lane), accK[nt]);
      accV[nt] = wmma_bf16(a, load_frag_b(Wb + 2 * WPLANE, XS, nt * 16, 0, lane), accV[nt]);
    }
    __syncthreads();                            // done reading before overwrite
  }

  // q/k: direct stores (half-wave writes 32B segments).
  const size_t head = (size_t)b * H_ + h;
#pragma unroll
  for (int nt = 0; nt < 4; ++nt) {
#pragma unroll
    for (int r = 0; r < 8; ++r) {
      const int m = r + 8 * half, n = nt * 16 + nl;
      const int t = t0 + wave * 16 + m;
      const size_t qk = (head * T_ + t) * D_ + n;
      qO[qk] = (__bf16)accQ[nt][r];
      kO[qk] = (__bf16)accK[nt][r];
    }
  }

  // v: transpose through LDS overlay, then coalesced 16B stores.
#pragma unroll
  for (int nt = 0; nt < 4; ++nt)
#pragma unroll
    for (int r = 0; r < 8; ++r) {
      const int m = r + 8 * half, n = nt * 16 + nl;
      ldsVt[n * 136 + wave * 16 + m] = (__bf16)accV[nt][r];  // [d][t_local]
    }
  __syncthreads();
  for (int i = tid; i < 1024; i += 256) {
    const int d = i >> 4, c8 = (i & 15) * 8;
    *(bf16x8*)&vTO[(head * D_ + d) * T_ + t0 + c8] =
        *(const bf16x8*)&ldsVt[d * 136 + c8];
  }
}

// ===========================================================================
// Kernel 2: causal flash attention with TDM-double-buffered K/V blocks.
// Grid (T/128, H, B), 256 threads; wave w owns 16 query rows; key blocks 64.
// ===========================================================================
#define KS 72                     // 64 + 4-DWORD TDM pad
#define KTILE (64 * KS)
#define PS 72

__global__ __launch_bounds__(256)
void flash_attn_kernel(const __bf16* __restrict__ q,
                       const __bf16* __restrict__ k,
                       const __bf16* __restrict__ vT,
                       float* __restrict__ out) {
  const int h = blockIdx.y, b = blockIdx.z;
  const int qb = blockIdx.x * 128;
  const int tid = threadIdx.x;
  const int wave = tid >> 5, lane = tid & 31;
  const int half = lane >> 4, nl = lane & 15;

  __shared__ __bf16 ldsK[2 * KTILE];     // [key][d], K=d contiguous
  __shared__ __bf16 ldsV[2 * KTILE];     // [d][key], K=key contiguous
  __shared__ __bf16 ldsP[8][16 * PS];    // per-wave P staging

  const size_t head = (size_t)b * H_ + h;
  const __bf16* qh = q  + head * T_ * D_;
  const __bf16* kh = k  + head * T_ * D_;
  const __bf16* vh = vT + head * D_ * T_;

  const bf16x16 aq0 = load_frag_a(qh + (size_t)(qb + wave * 16) * D_, D_, 0, 0,  lane);
  const bf16x16 aq1 = load_frag_a(qh + (size_t)(qb + wave * 16) * D_, D_, 0, 32, lane);

  f32x8 accO[4];
#pragma unroll
  for (int i = 0; i < 4; ++i) accO[i] = (f32x8)(0.0f);
  float m_i[8], l_i[8];
#pragma unroll
  for (int r = 0; r < 8; ++r) { m_i[r] = -1.0e30f; l_i[r] = 0.0f; }

  const float scale = 0.125f;
  const int nb = (qb + 128) / 64;        // causal block count (uniform in WG)

  if (wave == 0) {                       // prologue DMA: block 0 -> buffer 0
    tdm_load(lds_off_of(ldsK), kh, 64, 64, 0, D_, 0, /*padI=*/4, /*padA=*/3);
    tdm_load(lds_off_of(ldsV), vh, 64, 64, 0, T_, 0, 4, 3);
  }

  for (int ib = 0; ib < nb; ++ib) {
    const int s0 = ib * 64;
    if (wave == 0) {
      if (ib + 1 < nb) {                 // prefetch next block into other buf
        const int s1 = s0 + 64, nbuf = (ib + 1) & 1;
        tdm_load(lds_off_of(ldsK + nbuf * KTILE), kh + (size_t)s1 * D_,
                 64, 64, 0, D_, 0, 4, 3);
        tdm_load(lds_off_of(ldsV + nbuf * KTILE), vh + s1,
                 64, 64, 0, T_, 0, 4, 3);
        __builtin_amdgcn_s_wait_tensorcnt(2);
      } else {
        __builtin_amdgcn_s_wait_tensorcnt(0);
      }
    }
    __syncthreads();                     // current buffers ready for all waves
    const __bf16* Kb = ldsK + (ib & 1) * KTILE;
    const __bf16* Vb = ldsV + (ib & 1) * KTILE;

    // ---- S = Q K^T -------------------------------------------------------
    f32x8 S[4];
#pragma unroll
    for (int nt = 0; nt < 4; ++nt) {
      S[nt] = (f32x8)(0.0f);
      S[nt] = wmma_bf16(aq0, load_frag_b(Kb, KS, nt * 16, 0,  lane), S[nt]);
      S[nt] = wmma_bf16(aq1, load_frag_b(Kb, KS, nt * 16, 32, lane), S[nt]);
    }

    // ---- scale + causal mask + online softmax ----------------------------
    float rm[8];
#pragma unroll
    for (int r = 0; r < 8; ++r) rm[r] = -1.0e30f;
#pragma unroll
    for (int nt = 0; nt < 4; ++nt)
#pragma unroll
      for (int r = 0; r < 8; ++r) {
        const int qrow = qb + wave * 16 + r + 8 * half;
        const int kcol = s0 + nt * 16 + nl;
        float v = S[nt][r] * scale;
        v = (kcol <= qrow) ? v : -1.0e30f;
        S[nt][r] = v;
        rm[r] = fmaxf(rm[r], v);
      }
#pragma unroll
    for (int r = 0; r < 8; ++r)
#pragma unroll
      for (int mk = 8; mk >= 1; mk >>= 1)
        rm[r] = fmaxf(rm[r], __shfl_xor(rm[r], mk, 32));

    float alpha[8], rs[8];
#pragma unroll
    for (int r = 0; r < 8; ++r) {
      const float mn = fmaxf(m_i[r], rm[r]);
      alpha[r] = __expf(m_i[r] - mn);
      m_i[r] = mn;
      rs[r] = 0.0f;
    }
#pragma unroll
    for (int nt = 0; nt < 4; ++nt)
#pragma unroll
      for (int r = 0; r < 8; ++r) {
        const float p = __expf(S[nt][r] - m_i[r]);
        S[nt][r] = p;
        rs[r] += p;
      }
#pragma unroll
    for (int r = 0; r < 8; ++r) {
#pragma unroll
      for (int mk = 8; mk >= 1; mk >>= 1)
        rs[r] += __shfl_xor(rs[r], mk, 32);
      l_i[r] = l_i[r] * alpha[r] + rs[r];
    }
#pragma unroll
    for (int nt = 0; nt < 4; ++nt)
#pragma unroll
      for (int r = 0; r < 8; ++r) accO[nt][r] *= alpha[r];

    // ---- P through per-wave LDS into A-operand layout --------------------
    __bf16* Pw = &ldsP[wave][0];
#pragma unroll
    for (int nt = 0; nt < 4; ++nt)
#pragma unroll
      for (int r = 0; r < 8; ++r)
        Pw[(r + 8 * half) * PS + nt * 16 + nl] = (__bf16)S[nt][r];
    asm volatile("s_wait_dscnt 0x0" ::: "memory");
    __builtin_amdgcn_wave_barrier();

    const bf16x16 ap0 = load_frag_a(Pw, PS, 0, 0,  lane);
    const bf16x16 ap1 = load_frag_a(Pw, PS, 0, 32, lane);
#pragma unroll
    for (int nt = 0; nt < 4; ++nt) {     // O += P V
      accO[nt] = wmma_bf16(ap0, load_frag_b(Vb, KS, nt * 16, 0,  lane), accO[nt]);
      accO[nt] = wmma_bf16(ap1, load_frag_b(Vb, KS, nt * 16, 32, lane), accO[nt]);
    }
    __syncthreads();                     // reads done before buf overwrite
  }

  // ---- epilogue ----------------------------------------------------------
#pragma unroll
  for (int nt = 0; nt < 4; ++nt)
#pragma unroll
    for (int r = 0; r < 8; ++r) {
      const int t = qb + wave * 16 + r + 8 * half;
      const int d = nt * 16 + nl;
      out[((size_t)b * T_ + t) * (H_ * D_) + h * D_ + d] = accO[nt][r] / l_i[r];
    }
}

// ===========================================================================
extern "C" void kernel_launch(void* const* d_in, const int* in_sizes, int n_in,
                              void* d_out, int out_size, void* d_ws, size_t ws_size,
                              hipStream_t stream) {
  const float* x  = (const float*)d_in[0];
  const float* Wq = (const float*)d_in[1];
  const float* Wk = (const float*)d_in[2];
  const float* Wv = (const float*)d_in[3];

  const size_t nX = (size_t)B_ * T_ * C_;            // 8.4M
  const size_t nW = (size_t)3 * H_ * D_ * C_;        // 3.1M
  const size_t nQ = (size_t)B_ * H_ * T_ * D_;       // 8.4M

  __bf16* xb = (__bf16*)d_ws;
  __bf16* wT = xb + nX;
  __bf16* qw = wT + nW;
  __bf16* kw = qw + nQ;
  __bf16* vw = kw + nQ;                              // [B,H,D,T]

  cvt_x_kernel<<<dim3((unsigned)(nX / (256 * 8))), 256, 0, stream>>>(x, xb);
  cvt_w_kernel<<<dim3(C_ / 64, H_, 3), 256, 0, stream>>>(Wq, Wk, Wv, wT);
  qkv_proj_kernel<<<dim3(T_ / 128, H_, B_), 256, 0, stream>>>(xb, wT, qw, kw, vw);
  flash_attn_kernel<<<dim3(T_ / 128, H_, B_), 256, 0, stream>>>(qw, kw, vw,
                                                                (float*)d_out);
}
